// KernelAggregation_17059610100222
// MI455X (gfx1250) — matrix operand order repaired and checked by reading, not compiled, not run on hardware
//
#include <hip/hip_runtime.h>

// ---------------------------------------------------------------------------
// Dynamic conv (CondConv): per-sample mixed 3x3, 64->64 ch, 128x128, SAME pad.
// Implicit GEMM on v_wmma_f32_16x16x32_f16 (wave32, gfx1250).
//   M = cout (64), N = pixels, K = 9 taps x 64 ci = 576, tap-outer/ci-inner.
// WG tile: 64 cout x (4 rows x 16 cols) pixels; halo patch 6x18x64ch in LDS.
// ---------------------------------------------------------------------------

typedef __attribute__((ext_vector_type(16))) _Float16 v16h;
typedef __attribute__((ext_vector_type(8)))  _Float16 v8h;
typedef __attribute__((ext_vector_type(8)))  float    v8f;

#define BATCH   32
#define DIM     64
#define KS      3
#define NK      4
#define IH      128
#define IW      128
#define KDIM    (DIM * KS * KS)   // 576
#define TILE_H  4
#define TILE_WW 16
#define NHB     (IH / TILE_H)     // 32 row-blocks
#define NWB     (IW / TILE_WW)    // 8 col-blocks
#define PROWS   (TILE_H + 2)      // 6 patch rows
#define PCOLS   (TILE_WW + 2)     // 18 patch cols
#define LDSC    72                // padded ci stride in LDS (144B, 16B aligned)

// ------------------------- Stage 1: weight mixing --------------------------
// A[b][m][k] (f16), k = (kh*3+kw)*64 + ci.  weight is [NK][DIM][DIM][KS][KS].
__global__ void mix_weights_kernel(const float* __restrict__ att,
                                   const float* __restrict__ weight,
                                   _Float16* __restrict__ A) {
    int idx = blockIdx.x * blockDim.x + threadIdx.x;
    if (idx >= BATCH * DIM * KDIM) return;
    int k  = idx % KDIM;
    int m  = (idx / KDIM) % DIM;
    int b  = idx / (KDIM * DIM);
    int ci = k & 63;
    int r  = k >> 6;            // 0..8
    int kh = r / KS, kw = r % KS;
    int wofs = ((m * DIM + ci) * KS + kh) * KS + kw;
    float s = 0.f;
#pragma unroll
    for (int j = 0; j < NK; ++j)
        s += att[b * NK + j] * weight[j * (DIM * DIM * KS * KS) + wofs];
    A[idx] = (_Float16)s;
}

__global__ void mix_bias_kernel(const float* __restrict__ att,
                                const float* __restrict__ bias_p,
                                float* __restrict__ bmix) {
    int idx = blockIdx.x * blockDim.x + threadIdx.x;
    if (idx >= BATCH * DIM) return;
    int m = idx % DIM;
    int b = idx / DIM;
    float s = 0.f;
#pragma unroll
    for (int j = 0; j < NK; ++j)
        s += att[b * NK + j] * bias_p[j * DIM + m];
    bmix[idx] = s;
}

// ------------------------- Stage 2: WMMA conv ------------------------------
// Grid: BATCH * NHB * NWB workgroups of 256 threads (8 waves).
// WG computes out[b, 0:64, h0:h0+4, w0:w0+16].
__global__ __launch_bounds__(256)
void conv_wmma_kernel(const float* __restrict__ x,
                      const _Float16* __restrict__ A,
                      const float* __restrict__ bmix,
                      float* __restrict__ out) {
    // Halo patch, channel-innermost: xs[(dh*PCOLS + col)*LDSC + ci]
    __shared__ __align__(16) _Float16 xs[PROWS * PCOLS * LDSC];  // 15552 B

    const int wg  = blockIdx.x;
    const int wb  = wg % NWB;
    const int hb  = (wg / NWB) % NHB;
    const int b   = wg / (NWB * NHB);
    const int h0  = hb * TILE_H;
    const int w0  = wb * TILE_WW;

    const int tid = threadIdx.x;

    // ---- cooperative LDS fill (f32 -> f16, zero pad out-of-image) ----
    // (dh, ci) x col, col fastest; 6*64*18 = 6912 = 27 * 256 exactly.
    const int NFILL = PROWS * DIM * PCOLS;
    for (int i = tid; i < NFILL; i += 256) {
        int col = i % PCOLS;
        int t   = i / PCOLS;
        int ci  = t % DIM;
        int dh  = t / DIM;
        int hh  = h0 - 1 + dh;
        int ww  = w0 - 1 + col;
        float v = 0.f;
        if ((unsigned)hh < IH && (unsigned)ww < IW)
            v = x[((b * DIM + ci) * IH + hh) * IW + ww];
        xs[(dh * PCOLS + col) * LDSC + ci] = (_Float16)v;
    }
    __syncthreads();

    const int lane   = tid & 31;
    const int wave   = tid >> 5;
    const int m_tile = wave >> 1;       // 0..3 : cout tile
    const int n_pair = wave & 1;        // 0..1 : pixel-row pair
    const int lrow   = lane & 15;       // N column within tile (= w offset)
    const int lhalf  = lane >> 4;

    const int m = m_tile * 16 + lrow;   // A row for this lane (both K-halves)
    const _Float16* __restrict__ Arow = A + (b * DIM + m) * KDIM;

    v8f acc0 = {};   // output row r0 = n_pair*2 + 0
    v8f acc1 = {};   // output row r0 = n_pair*2 + 1

#pragma unroll
    for (int r = 0; r < 9; ++r) {
        const int kh = r / KS;
        const int kw = r % KS;
#pragma unroll
        for (int ch = 0; ch < 2; ++ch) {
            const int k0 = r * 64 + ch * 32;

            // A fragment (global, L2-resident): 16 f16 per lane.
            // lane<16: K = k0+{0..15}; lane>=16: K = k0+{16..31}
            union { v16h v; v8h h2[2]; } af;
            af.h2[0] = *(const v8h*)(Arow + k0 + lhalf * 16);
            af.h2[1] = *(const v8h*)(Arow + k0 + 8 + lhalf * 16);

            // B fragments (LDS): lane column N = lrow, K-half by lhalf,
            // 16 contiguous f16 along ci. Output row r0 selects patch row.
#pragma unroll
            for (int ns = 0; ns < 2; ++ns) {
                const int r0 = n_pair * 2 + ns;                 // 0..3
                const _Float16* bp =
                    &xs[((r0 + kh) * PCOLS + lrow + kw) * LDSC + ch * 32 + lhalf * 16];
                v16h bf = *(const v16h*)bp;
                if (ns == 0)
                    acc0 = __builtin_amdgcn_wmma_f32_16x16x32_f16(
                        false, af.v, false, bf, (short)0, acc0, false, false);
                else
                    acc1 = __builtin_amdgcn_wmma_f32_16x16x32_f16(
                        false, af.v, false, bf, (short)0, acc1, false, false);
            }
        }
    }

    // ---- epilogue: bias + store (C/D layout: VGPR v -> M = v + 8*lhalf) ----
    float bias[8];
#pragma unroll
    for (int v = 0; v < 8; ++v)
        bias[v] = bmix[b * DIM + m_tile * 16 + v + 8 * lhalf];

#pragma unroll
    for (int ns = 0; ns < 2; ++ns) {
        const int r0 = n_pair * 2 + ns;
        const int hh = h0 + r0;
#pragma unroll
        for (int v = 0; v < 8; ++v) {
            const int mm = m_tile * 16 + v + 8 * lhalf;
            const float val = (ns == 0 ? acc0[v] : acc1[v]) + bias[v];
            out[((b * DIM + mm) * IH + hh) * IW + w0 + lrow] = val;
        }
    }
}

// ---------------------------------------------------------------------------
extern "C" void kernel_launch(void* const* d_in, const int* in_sizes, int n_in,
                              void* d_out, int out_size, void* d_ws, size_t ws_size,
                              hipStream_t stream) {
    const float* x      = (const float*)d_in[0];   // [32,64,128,128]
    const float* att    = (const float*)d_in[1];   // [32,4]
    const float* weight = (const float*)d_in[2];   // [4,64,64,3,3]
    const float* bias_p = (const float*)d_in[3];   // [4,64]

    _Float16* A   = (_Float16*)d_ws;                                   // 2.36 MB
    float*    bmx = (float*)((char*)d_ws + (size_t)BATCH * DIM * KDIM * sizeof(_Float16));
    float*    out = (float*)d_out;

    const int nA = BATCH * DIM * KDIM;
    mix_weights_kernel<<<(nA + 255) / 256, 256, 0, stream>>>(att, weight, A);
    mix_bias_kernel<<<(BATCH * DIM + 255) / 256, 256, 0, stream>>>(att, bias_p, bmx);
    conv_wmma_kernel<<<BATCH * NHB * NWB, 256, 0, stream>>>(x, A, bmx, out);
}
